// TransformerBlock_77283641524433
// MI455X (gfx1250) — compile-verified, matrix operand order checked by reading, not run on hardware
//
#include <hip/hip_runtime.h>
#include <hip/hip_bf16.h>
#include <math.h>
#include <stdint.h>

// ---------------------------------------------------------------------------
// Types for CDNA5 WMMA (wave32, v_wmma_f32_16x16x32_bf16)
// ---------------------------------------------------------------------------
typedef __attribute__((ext_vector_type(16))) __bf16 v16bf;
typedef __attribute__((ext_vector_type(8)))  float  v8f;
typedef __attribute__((ext_vector_type(8)))  unsigned short ushort8;
typedef __attribute__((ext_vector_type(4)))  int    v4i;

#define EMBED   1024
#define HIDDEN  4096
#define SEQ     2048
#define BATCH   2
#define HEADS   16
#define HDIM    64
#define MROWS   (BATCH*SEQ)   // 4096

// ---------------------------------------------------------------------------
// Async global -> LDS copy (CDNA5 GLOBAL_LOAD_ASYNC_TO_LDS_B128, ASYNCcnt).
// ---------------------------------------------------------------------------
#if defined(__HIP_DEVICE_COMPILE__) && __has_builtin(__builtin_amdgcn_global_load_async_to_lds_b128)
#define HAVE_ASYNC_LDS 1
#else
#define HAVE_ASYNC_LDS 0
#endif

__device__ __forceinline__ void cp16(__bf16* ldsPtr, const __bf16* gPtr) {
#if HAVE_ASYNC_LDS
  __builtin_amdgcn_global_load_async_to_lds_b128(
      (__attribute__((address_space(1))) v4i*)(uintptr_t)gPtr,
      (__attribute__((address_space(3))) v4i*)ldsPtr,
      0, 0);
#else
  *(ushort8*)ldsPtr = *(const ushort8*)gPtr;
#endif
}

__device__ __forceinline__ void wait_async_all() {
#if HAVE_ASYNC_LDS
#if __has_builtin(__builtin_amdgcn_s_wait_asynccnt)
  __builtin_amdgcn_s_wait_asynccnt(0);
#else
  asm volatile("s_wait_asynccnt 0x0" ::: "memory");
#endif
#endif
}

// ---------------------------------------------------------------------------
// WMMA fragment helpers (layouts per CDNA5 ISA 7.12.2):
//   lane = half*16 + idx ; element e -> K(e) = half*8 + (e<8 ? e : e+8)
// -> elements 0..7 and 8..15 are each 16 contiguous bytes (two b128 loads).
// ---------------------------------------------------------------------------
__device__ __forceinline__ v16bf frag_rowmajor(const __bf16* p, int ld, int row, int col) {
  int lane = threadIdx.x & 31;
  int half = lane >> 4, idx = lane & 15;
  const __bf16* r = p + (size_t)(row + idx) * ld + col + half * 8;
  v16bf f;
  ((ushort8*)&f)[0] = *(const ushort8*)(r);
  ((ushort8*)&f)[1] = *(const ushort8*)(r + 16);
  return f;
}

// B-fragment where K runs down rows of a row-major [K x N] source (strided).
__device__ __forceinline__ v16bf frag_b_strided(const __bf16* p, int ld, int krow0, int col0) {
  int lane = threadIdx.x & 31;
  int half = lane >> 4, idx = lane & 15;
  v16bf f;
  __bf16* e = (__bf16*)&f;
  int kb = krow0 + half * 8;
#pragma unroll
  for (int t = 0; t < 8; ++t) e[t]     = p[(size_t)(kb + t)      * ld + col0 + idx];
#pragma unroll
  for (int t = 0; t < 8; ++t) e[8 + t] = p[(size_t)(kb + 16 + t) * ld + col0 + idx];
  return f;
}

__device__ __forceinline__ v8f wmma_bf16(v16bf a, v16bf b, v8f c) {
  return __builtin_amdgcn_wmma_f32_16x16x32_bf16(false, a, false, b, (short)0, c, false, false);
}

// ---------------------------------------------------------------------------
// Weight prep: fp32 [K x N] -> bf16 transposed [N x K] (once per launch).
// ---------------------------------------------------------------------------
__global__ __launch_bounds__(256) void cvt_transpose_bf16(const float* __restrict__ W,
                                                          __bf16* __restrict__ Bt,
                                                          int K, int N) {
  __shared__ float t[32][33];
  int n0 = blockIdx.x * 32, k0 = blockIdx.y * 32;
  int tx = threadIdx.x & 31, ty = threadIdx.x >> 5;  // ty: 0..7
#pragma unroll
  for (int i = 0; i < 32; i += 8)
    t[ty + i][tx] = W[(size_t)(k0 + ty + i) * N + n0 + tx];
  __syncthreads();
#pragma unroll
  for (int i = 0; i < 32; i += 8)
    Bt[(size_t)(n0 + ty + i) * K + k0 + tx] = (__bf16)t[tx][ty + i];
}

// ---------------------------------------------------------------------------
// LayerNorm over 1024 columns -> bf16 output. One block (256 thr) per row.
// ---------------------------------------------------------------------------
__global__ __launch_bounds__(256) void ln_bf16_kernel(const float* __restrict__ x,
                                                      const float* __restrict__ w,
                                                      const float* __restrict__ b,
                                                      __bf16* __restrict__ out) {
  int row = blockIdx.x;
  const float* xr = x + (size_t)row * EMBED;
  float s = 0.f, ss = 0.f;
  for (int i = threadIdx.x; i < EMBED; i += 256) {
    float v = xr[i]; s += v; ss += v * v;
  }
#pragma unroll
  for (int m = 1; m < 32; m <<= 1) {
    s  += __shfl_xor(s,  m, 32);
    ss += __shfl_xor(ss, m, 32);
  }
  __shared__ float rs[8], rss[8];
  int wv = threadIdx.x >> 5, ln = threadIdx.x & 31;
  if (ln == 0) { rs[wv] = s; rss[wv] = ss; }
  __syncthreads();
  if (threadIdx.x == 0) {
    float a = 0.f, c = 0.f;
#pragma unroll
    for (int i = 0; i < 8; ++i) { a += rs[i]; c += rss[i]; }
    rs[0] = a; rss[0] = c;
  }
  __syncthreads();
  float mean = rs[0] * (1.f / EMBED);
  float var  = rss[0] * (1.f / EMBED) - mean * mean;
  float inv  = rsqrtf(var + 1e-5f);
  __bf16* orow = out + (size_t)row * EMBED;
  for (int i = threadIdx.x; i < EMBED; i += 256)
    orow[i] = (__bf16)((xr[i] - mean) * inv * w[i] + b[i]);
}

// ---------------------------------------------------------------------------
// Tiled WMMA GEMM:  out = epilogue(A[M,K]bf16 @ Bt[N,K]bf16^T + bias)
// Async-staged, double-buffered LDS. BM=BN=128, BK=64, 8 waves, 64x32/wave.
// ---------------------------------------------------------------------------
#define BM 128
#define BN 128
#define BK 64
#define LDP 80   // padded K stride in LDS elements (160B, 16B aligned rows)

__device__ __forceinline__ void stage_tile(__bf16 (*dst)[LDP], const __bf16* src,
                                           int ld, int row0, int k0, int tid) {
#pragma unroll
  for (int c = 0; c < 4; ++c) {
    int u = c * 256 + tid;            // 0..1023 : 128 rows x 8 chunks
    int r = u >> 3, kc = (u & 7) * 8;
    cp16(&dst[r][kc], src + (size_t)(row0 + r) * ld + k0 + kc);
  }
}

template <bool GELU>
__global__ __launch_bounds__(256, 2) void gemm_wmma(const __bf16* __restrict__ A,
                                                    const __bf16* __restrict__ Bt,
                                                    const float* __restrict__ bias,
                                                    const float* __restrict__ resid,
                                                    float* __restrict__ outF,
                                                    __bf16* __restrict__ outB,
                                                    int M, int N, int K) {
  __shared__ __attribute__((aligned(16))) __bf16 As[2][BM][LDP];
  __shared__ __attribute__((aligned(16))) __bf16 Bs[2][BN][LDP];

  int bm0 = blockIdx.y * BM, bn0 = blockIdx.x * BN;
  int tid = threadIdx.x;
  int wv = tid >> 5, lane = tid & 31, half = lane >> 4, idx = lane & 15;
  int wm = (wv >> 2) * 64, wn = (wv & 3) * 32;

  v8f acc[4][2];
#pragma unroll
  for (int mi = 0; mi < 4; ++mi)
#pragma unroll
    for (int ni = 0; ni < 2; ++ni) { v8f z = {}; acc[mi][ni] = z; }

  stage_tile(As[0], A,  K, bm0, 0, tid);
  stage_tile(Bs[0], Bt, K, bn0, 0, tid);
  wait_async_all();
  __syncthreads();

  int nk = K / BK;
  for (int it = 0; it < nk; ++it) {
    int cur = it & 1;
    if (it + 1 < nk) {
      stage_tile(As[cur ^ 1], A,  K, bm0, (it + 1) * BK, tid);
      stage_tile(Bs[cur ^ 1], Bt, K, bn0, (it + 1) * BK, tid);
    }
#pragma unroll
    for (int kk = 0; kk < BK; kk += 32) {
      v16bf af[4], bf2[2];
#pragma unroll
      for (int mi = 0; mi < 4; ++mi)
        af[mi] = frag_rowmajor(&As[cur][0][0], LDP, wm + mi * 16, kk);
#pragma unroll
      for (int ni = 0; ni < 2; ++ni)
        bf2[ni] = frag_rowmajor(&Bs[cur][0][0], LDP, wn + ni * 16, kk);
#pragma unroll
      for (int mi = 0; mi < 4; ++mi)
#pragma unroll
        for (int ni = 0; ni < 2; ++ni)
          acc[mi][ni] = wmma_bf16(af[mi], bf2[ni], acc[mi][ni]);
    }
    wait_async_all();
    __syncthreads();
  }

  // --- epilogue: C layout row = j + half*8, col = idx
#pragma unroll
  for (int mi = 0; mi < 4; ++mi)
#pragma unroll
    for (int ni = 0; ni < 2; ++ni) {
      int col = bn0 + wn + ni * 16 + idx;
      float bv = bias ? bias[col] : 0.f;
#pragma unroll
      for (int j = 0; j < 8; ++j) {
        int row = bm0 + wm + mi * 16 + half * 8 + j;
        float v = acc[mi][ni][j] + bv;
        if (GELU) v = 0.5f * v * (1.f + erff(v * 0.70710678118654752f));
        if (resid) v += resid[(size_t)row * N + col];
        if (outF) outF[(size_t)row * N + col] = v;
        if (outB) outB[(size_t)row * N + col] = (__bf16)v;
      }
    }
}

// ---------------------------------------------------------------------------
// Flash attention (bf16 in/out), LDS-staged K/V shared by all 8 waves.
// Block = 256 thr = 8 waves, 128 queries of one (b,h); wave owns 16 queries.
// 64-key K/V tiles are async-DMA'd into double-buffered LDS; the next tile's
// DMA overlaps this tile's WMMA + softmax. Steady-state loop has no global
// loads: K-fragments are batched ds_load_b128, V-fragments LDS gathers.
// ---------------------------------------------------------------------------
#define KTILE 64
#define FLDP  72   // LDS K/V row stride (144B, 16B aligned)

__device__ __forceinline__ void stage_kv(__bf16 (*dst)[FLDP], const __bf16* src,
                                         int key0, int hbase, int tid) {
#pragma unroll
  for (int c = 0; c < 2; ++c) {
    int u = c * 256 + tid;            // 0..511 : 64 rows x 8 chunks
    int r = u >> 3, kc = (u & 7) * 8;
    cp16(&dst[r][kc], src + (size_t)(key0 + r) * EMBED + hbase + kc);
  }
}

__global__ __launch_bounds__(256, 2) void flash_attn(const __bf16* __restrict__ Q,
                                                     const __bf16* __restrict__ Km,
                                                     const __bf16* __restrict__ V,
                                                     __bf16* __restrict__ O) {
  int tid = threadIdx.x;
  int w = tid >> 5, lane = tid & 31;
  int half = lane >> 4, idx = lane & 15;
  int bh = blockIdx.y;
  int b = bh >> 4, h = bh & 15;
  int hbase = h * HDIM;
  int q0 = blockIdx.x * 128 + w * 16;
  size_t base = (size_t)b * SEQ * EMBED;
  const __bf16* Qp = Q + base;
  const __bf16* Kp = Km + base;
  const __bf16* Vp = V + base;

  __shared__ __attribute__((aligned(16))) __bf16 Kl[2][KTILE][FLDP];
  __shared__ __attribute__((aligned(16))) __bf16 Vl[2][KTILE][FLDP];
  __shared__ __attribute__((aligned(16))) __bf16 Pl[8][16][FLDP];

  // Q fragments (registers, loaded once) for the two 32-wide head-dim steps
  v16bf aQ0 = frag_rowmajor(Qp, EMBED, q0, hbase);
  v16bf aQ1 = frag_rowmajor(Qp, EMBED, q0, hbase + 32);

  v8f o[4];
#pragma unroll
  for (int of = 0; of < 4; ++of) { v8f z = {}; o[of] = z; }
  float mrow[8], lrow[8];
#pragma unroll
  for (int j = 0; j < 8; ++j) { mrow[j] = -1e30f; lrow[j] = 0.f; }

  const float scale = 0.125f;  // 1/sqrt(64)

  stage_kv(Kl[0], Kp, 0, hbase, tid);
  stage_kv(Vl[0], Vp, 0, hbase, tid);
  wait_async_all();
  __syncthreads();

  const int NT = SEQ / KTILE;
  for (int kt = 0; kt < NT; ++kt) {
    int cur = kt & 1;
    if (kt + 1 < NT) {
      stage_kv(Kl[cur ^ 1], Kp, (kt + 1) * KTILE, hbase, tid);
      stage_kv(Vl[cur ^ 1], Vp, (kt + 1) * KTILE, hbase, tid);
    }

    // ---- S = Q.K^T from LDS (batched ds_load_b128 fragments)
    const __bf16* Klp = (const __bf16*)&Kl[cur][0][0];
    v16bf kf[4][2];
#pragma unroll
    for (int nf = 0; nf < 4; ++nf) {
      kf[nf][0] = frag_rowmajor(Klp, FLDP, nf * 16, 0);
      kf[nf][1] = frag_rowmajor(Klp, FLDP, nf * 16, 32);
    }
    v8f s[4];
#pragma unroll
    for (int nf = 0; nf < 4; ++nf) {
      v8f sv = {};
      sv = wmma_bf16(aQ0, kf[nf][0], sv);
      sv = wmma_bf16(aQ1, kf[nf][1], sv);
      s[nf] = sv;
    }

    // ---- V fragments from LDS (gather), overlap softmax VALU below
    const __bf16* Vlp = (const __bf16*)&Vl[cur][0][0];
    v16bf vf[2][4];
#pragma unroll
    for (int s2 = 0; s2 < 2; ++s2)
#pragma unroll
      for (int of = 0; of < 4; ++of)
        vf[s2][of] = frag_b_strided(Vlp, FLDP, s2 * 32, of * 16);

    // ---- online softmax; row j lives on lanes of one half (masks <= 8)
#pragma unroll
    for (int j = 0; j < 8; ++j) {
      float v0 = fmaxf(fmaxf(s[0][j], s[1][j]), fmaxf(s[2][j], s[3][j]));
#pragma unroll
      for (int m = 1; m < 16; m <<= 1) v0 = fmaxf(v0, __shfl_xor(v0, m, 32));
      float mnew = fmaxf(mrow[j], v0 * scale);
      float alpha = __expf(mrow[j] - mnew);
      float rsum = 0.f;
#pragma unroll
      for (int nf = 0; nf < 4; ++nf) {
        float p = __expf(s[nf][j] * scale - mnew);
        Pl[w][half * 8 + j][nf * 16 + idx] = (__bf16)p;
        rsum += p;
      }
#pragma unroll
      for (int m = 1; m < 16; m <<= 1) rsum += __shfl_xor(rsum, m, 32);
      lrow[j] = lrow[j] * alpha + rsum;
      mrow[j] = mnew;
#pragma unroll
      for (int of = 0; of < 4; ++of) o[of][j] *= alpha;
    }

    // ---- O += P.V  (P restaged via wave-local LDS into A-fragments)
    const __bf16* Pw = (const __bf16*)&Pl[w][0][0];
#pragma unroll
    for (int s2 = 0; s2 < 2; ++s2) {
      v16bf aP = frag_rowmajor(Pw, FLDP, 0, s2 * 32);
#pragma unroll
      for (int of = 0; of < 4; ++of)
        o[of] = wmma_bf16(aP, vf[s2][of], o[of]);
    }

    wait_async_all();
    __syncthreads();
  }

  // ---- normalize and write [B,N,H,D]
#pragma unroll
  for (int of = 0; of < 4; ++of)
#pragma unroll
    for (int j = 0; j < 8; ++j) {
      int row = q0 + half * 8 + j;
      O[base + (size_t)row * EMBED + hbase + of * 16 + idx] =
          (__bf16)(o[of][j] / lrow[j]);
    }
}

// ---------------------------------------------------------------------------
// Host orchestration
// ---------------------------------------------------------------------------
extern "C" void kernel_launch(void* const* d_in, const int* in_sizes, int n_in,
                              void* d_out, int out_size, void* d_ws, size_t ws_size,
                              hipStream_t stream) {
  (void)in_sizes; (void)n_in; (void)out_size; (void)ws_size;

  const float* x     = (const float*)d_in[0];
  const float* ln1_w = (const float*)d_in[1];
  const float* ln1_b = (const float*)d_in[2];
  const float* ln2_w = (const float*)d_in[3];
  const float* ln2_b = (const float*)d_in[4];
  const float* wq = (const float*)d_in[5];
  const float* bq = (const float*)d_in[6];
  const float* wk = (const float*)d_in[7];
  const float* bk = (const float*)d_in[8];
  const float* wv = (const float*)d_in[9];
  const float* bv = (const float*)d_in[10];
  const float* wo = (const float*)d_in[11];
  const float* bo = (const float*)d_in[12];
  const float* w1 = (const float*)d_in[13];
  const float* b1 = (const float*)d_in[14];
  const float* w2 = (const float*)d_in[15];
  const float* b2 = (const float*)d_in[16];
  float* out = (float*)d_out;

  char* ws = (char*)d_ws;
  size_t off = 0;
  auto alloc = [&](size_t bytes) -> char* {
    char* p = ws + off;
    off = (off + bytes + 255) & ~(size_t)255;
    return p;
  };
  const size_t DD = (size_t)EMBED * EMBED;
  const size_t DH = (size_t)EMBED * HIDDEN;
  const size_t MD = (size_t)MROWS * EMBED;
  const size_t MH = (size_t)MROWS * HIDDEN;

  // transposed bf16 weights: Wt[N][K]
  __bf16* wq_t  = (__bf16*)alloc(DD * 2);
  __bf16* wk_t  = (__bf16*)alloc(DD * 2);
  __bf16* wv_t  = (__bf16*)alloc(DD * 2);
  __bf16* wo_t  = (__bf16*)alloc(DD * 2);
  __bf16* w1_t  = (__bf16*)alloc(DH * 2);
  __bf16* w2_t  = (__bf16*)alloc(DH * 2);
  __bf16* ln1o  = (__bf16*)alloc(MD * 2);
  __bf16* q_b   = (__bf16*)alloc(MD * 2);
  __bf16* k_b   = (__bf16*)alloc(MD * 2);
  __bf16* v_b   = (__bf16*)alloc(MD * 2);
  __bf16* attn_o= (__bf16*)alloc(MD * 2);
  float*  x1    = (float*)alloc(MD * 4);
  __bf16* ln2o  = (__bf16*)alloc(MD * 2);
  __bf16* h1    = (__bf16*)alloc(MH * 2);

  // 1) weights -> bf16, transposed to [N][K]
  dim3 tdd(EMBED / 32, EMBED / 32);
  cvt_transpose_bf16<<<tdd, 256, 0, stream>>>(wq, wq_t, EMBED, EMBED);
  cvt_transpose_bf16<<<tdd, 256, 0, stream>>>(wk, wk_t, EMBED, EMBED);
  cvt_transpose_bf16<<<tdd, 256, 0, stream>>>(wv, wv_t, EMBED, EMBED);
  cvt_transpose_bf16<<<tdd, 256, 0, stream>>>(wo, wo_t, EMBED, EMBED);
  cvt_transpose_bf16<<<dim3(HIDDEN / 32, EMBED / 32), 256, 0, stream>>>(w1, w1_t, EMBED, HIDDEN);
  cvt_transpose_bf16<<<dim3(EMBED / 32, HIDDEN / 32), 256, 0, stream>>>(w2, w2_t, HIDDEN, EMBED);

  // 2) LN1
  ln_bf16_kernel<<<MROWS, 256, 0, stream>>>(x, ln1_w, ln1_b, ln1o);

  // 3) Q/K/V projections
  dim3 gqkv(EMBED / BN, MROWS / BM);
  gemm_wmma<false><<<gqkv, 256, 0, stream>>>(ln1o, wq_t, bq, nullptr, nullptr, q_b,
                                             MROWS, EMBED, EMBED);
  gemm_wmma<false><<<gqkv, 256, 0, stream>>>(ln1o, wk_t, bk, nullptr, nullptr, k_b,
                                             MROWS, EMBED, EMBED);
  gemm_wmma<false><<<gqkv, 256, 0, stream>>>(ln1o, wv_t, bv, nullptr, nullptr, v_b,
                                             MROWS, EMBED, EMBED);

  // 4) flash attention (128 queries per block, K/V LDS-shared by 8 waves)
  dim3 ga(SEQ / 128, BATCH * HEADS);
  flash_attn<<<ga, 256, 0, stream>>>(q_b, k_b, v_b, attn_o);

  // 5) output projection + residual -> x1 (f32)
  gemm_wmma<false><<<gqkv, 256, 0, stream>>>(attn_o, wo_t, bo, x, x1, nullptr,
                                             MROWS, EMBED, EMBED);

  // 6) LN2
  ln_bf16_kernel<<<MROWS, 256, 0, stream>>>(x1, ln2_w, ln2_b, ln2o);

  // 7) MLP up + exact GELU -> h1 (bf16)
  dim3 gm1(HIDDEN / BN, MROWS / BM);
  gemm_wmma<true><<<gm1, 256, 0, stream>>>(ln2o, w1_t, b1, nullptr, nullptr, h1,
                                           MROWS, HIDDEN, EMBED);

  // 8) MLP down + residual -> out (f32)
  dim3 gm2(EMBED / BN, MROWS / BM);
  gemm_wmma<false><<<gm2, 256, 0, stream>>>(h1, w2_t, b2, x1, out, nullptr,
                                            MROWS, EMBED, HIDDEN);
}
